// SmallthinkerAttention_62560493633736
// MI455X (gfx1250) — matrix operand order repaired and checked by reading, not compiled, not run on hardware
//
#include <hip/hip_runtime.h>
#include <hip/hip_bf16.h>

// ---------------------------------------------------------------------------
// SmallThinker attention for MI455X (gfx1250, wave32, WMMA + async LDS DMA).
// Pipeline:
//   1) cast fp32 -> bf16 (X, Wq, Wk, Wv, Wo) into workspace
//   2) gemm_nt (bf16 WMMA, fp32 acc, 128x128 tile, async double-buffered LDS):
//      Q = X·Wq^T, K = X·Wk^T, V = X·Wv^T
//   3) RoPE in-place on bf16 Q and K
//   4) attention: per (b, h, 16-row q tile): QK^T WMMA -> masked logits ->
//      softmax (writes mandatory fp32 attn_weights to d_out) -> P·V WMMA
//   5) gemm_nt: out = AO·Wo^T (fp32 epilogue to d_out)
// ---------------------------------------------------------------------------

typedef __bf16 bf16_t;
typedef __attribute__((ext_vector_type(16))) __bf16 v16bf;
typedef __attribute__((ext_vector_type(8)))  __bf16 v8bf;
typedef __attribute__((ext_vector_type(8)))  float  v8f;

#define SHUF16(a, b) __builtin_shufflevector(a, b, 0,1,2,3,4,5,6,7,8,9,10,11,12,13,14,15)

constexpr int B_   = 2;
constexpr int S_   = 2048;
constexpr int HID_ = 2048;
constexpr int NH_  = 32;
constexpr int NKV_ = 8;
constexpr int HD_  = 64;
constexpr float SCALING_ = 0.125f;   // 64^-0.5
constexpr float NEG_INF_ = -1e9f;

// --- gfx1250 async global->LDS copy (ASYNCcnt tracked) ----------------------
// VDST = per-lane LDS byte address, VADDR = per-lane 64-bit global address.
__device__ __forceinline__ void async_copy_b128(unsigned lds_off, const void* g) {
  asm volatile("global_load_async_to_lds_b128 %0, %1, off"
               :: "v"(lds_off), "v"(g) : "memory");
}
__device__ __forceinline__ void wait_async0() {
  asm volatile("s_wait_asynccnt 0x0" ::: "memory");
}
__device__ __forceinline__ void wait_ds0() {
  asm volatile("s_wait_dscnt 0x0" ::: "memory");
}
__device__ __forceinline__ unsigned lds_addr32(const void* p) {
  return (unsigned)(size_t)p;   // LDS aperture keeps byte offset in addr[31:0]
}

// ---------------------------------------------------------------------------
// fp32 -> bf16 cast
// ---------------------------------------------------------------------------
__global__ void cast_f32_bf16(const float* __restrict__ in,
                              bf16_t* __restrict__ out, int n) {
  int i = blockIdx.x * blockDim.x + threadIdx.x;
  if (i < n) out[i] = (bf16_t)in[i];
}

// ---------------------------------------------------------------------------
// C[m][n] = sum_k A[m][k] * Bm[n][k]   (both bf16 row-major, fp32 accumulate)
// 256 threads = 8 waves; block tile 128x128; wave tile 32x64; K step 64.
// Double-buffered LDS filled by async global->LDS DMA.
// ---------------------------------------------------------------------------
template <bool OUT_BF16>
__global__ __launch_bounds__(256)
void gemm_nt_bf16(const bf16_t* __restrict__ A, const bf16_t* __restrict__ Bm,
                  void* __restrict__ C, int M, int N, int K) {
  __shared__ alignas(16) bf16_t As[2][128][72];  // 72*2B stride: conflict-free
  __shared__ alignas(16) bf16_t Bs[2][128][72];

  const int tid   = threadIdx.x;
  const int lane  = tid & 31;
  const int wave  = tid >> 5;
  const int wm    = (wave & 3) * 32;   // wave row base in block tile
  const int wn    = (wave >> 2) * 64;  // wave col base in block tile
  const int m0    = blockIdx.x * 128;
  const int n0    = blockIdx.y * 128;
  const int mlane = lane & 15;
  const int hi    = lane >> 4;         // 0 | 1
  const int koffA = hi * 8;            // A frag: k chunks {koffA, 16+koffA}
  const int koffB = hi * 8;            // B frag: k chunks {koffB, 16+koffB}

  // staging: each thread owns 32 contiguous k of one row in A and one in B
  const int sr = tid >> 1;             // 0..127
  const int sh = (tid & 1) * 32;       // 0 | 32

  auto stage = [&](int buf, int k0) {
    const bf16_t* ga = A  + (size_t)(m0 + sr) * K + k0 + sh;
    const bf16_t* gb = Bm + (size_t)(n0 + sr) * K + k0 + sh;
    unsigned la = lds_addr32(&As[buf][sr][sh]);
    unsigned lb = lds_addr32(&Bs[buf][sr][sh]);
#pragma unroll
    for (int j = 0; j < 4; j++) {      // 4 x 16B per matrix per thread
      async_copy_b128(la + j * 16, ga + j * 8);
      async_copy_b128(lb + j * 16, gb + j * 8);
    }
  };

  v8f acc[2][4] = {};

  stage(0, 0);
  wait_async0();
  __syncthreads();

  int buf = 0;
  for (int k0 = 0; k0 < K; k0 += 64) {
    if (k0 + 64 < K) stage(buf ^ 1, k0 + 64);   // overlap DMA with compute

#pragma unroll
    for (int ks = 0; ks < 64; ks += 32) {
      v16bf afr[2], bfr[4];
#pragma unroll
      for (int t = 0; t < 2; t++) {
        const bf16_t* ap = &As[buf][wm + t * 16 + mlane][ks + koffA];
        afr[t] = SHUF16(*(const v8bf*)ap, *(const v8bf*)(ap + 16));
      }
#pragma unroll
      for (int t = 0; t < 4; t++) {
        const bf16_t* bp = &Bs[buf][wn + t * 16 + mlane][ks + koffB];
        bfr[t] = SHUF16(*(const v8bf*)bp, *(const v8bf*)(bp + 16));
      }
#pragma unroll
      for (int i = 0; i < 2; i++)
#pragma unroll
        for (int j = 0; j < 4; j++)
          acc[i][j] = __builtin_amdgcn_wmma_f32_16x16x32_bf16(
              false, afr[i], false, bfr[j], (short)0, acc[i][j], false, false);
    }

    wait_async0();
    __syncthreads();
    buf ^= 1;
  }

#pragma unroll
  for (int i = 0; i < 2; i++)
#pragma unroll
    for (int j = 0; j < 4; j++)
#pragma unroll
      for (int r = 0; r < 8; r++) {
        int m = m0 + wm + i * 16 + r + hi * 8;
        int n = n0 + wn + j * 16 + mlane;
        float v = acc[i][j][r];
        if (OUT_BF16) ((bf16_t*)C)[(size_t)m * N + n] = (bf16_t)v;
        else          ((float*)C)[(size_t)m * N + n]  = v;
      }
}

// ---------------------------------------------------------------------------
// RoPE in-place on bf16 [B*S][nheads*64]; one thread per (token, head, d<32)
// ---------------------------------------------------------------------------
__global__ void rope_kernel(bf16_t* __restrict__ QK,
                            const float* __restrict__ cosp,
                            const float* __restrict__ sinp, int nheads) {
  int idx = blockIdx.x * blockDim.x + threadIdx.x;
  int total = B_ * S_ * nheads * 32;
  if (idx >= total) return;
  int d = idx & 31;
  int h = (idx >> 5) % nheads;
  int t = idx / (32 * nheads);                 // token = b*S + s
  size_t base = (size_t)t * (nheads * 64) + h * 64;
  float x1 = (float)QK[base + d];
  float x2 = (float)QK[base + d + 32];
  size_t cb = (size_t)t * 64;
  float c1 = cosp[cb + d],      s1 = sinp[cb + d];
  float c2 = cosp[cb + d + 32], s2 = sinp[cb + d + 32];
  QK[base + d]      = (bf16_t)(x1 * c1 - x2 * s1);
  QK[base + d + 32] = (bf16_t)(x2 * c2 + x1 * s2);
}

// ---------------------------------------------------------------------------
// Attention core. 1 wave per (b, h, q-tile of 16 rows).
//  phase 1: masked logits (QK^T WMMA) -> attw region of d_out (L2 resident)
//  phase 2: two-pass softmax, writes fp32 attn_weights + zero upper triangle
//  phase 3: P·V WMMA, V staged via async LDS DMA, attn_out (bf16) to ws
// ---------------------------------------------------------------------------
__global__ __launch_bounds__(32)
void attn_kernel(const bf16_t* __restrict__ Q, const bf16_t* __restrict__ K,
                 const bf16_t* __restrict__ V, float* __restrict__ attw,
                 bf16_t* __restrict__ AO) {
  __shared__ alignas(16) bf16_t Vt[32][64];

  const int lane  = threadIdx.x;
  const int q0    = blockIdx.x * 16;
  const int h     = blockIdx.y;
  const int b     = blockIdx.z;
  const int kh    = h >> 2;            // GQA: kv head = h / (NH/NKV)
  const int mlane = lane & 15;
  const int hi    = lane >> 4;
  const int kend  = q0 + 16;           // causal: rows need k <= q0+15

  float* aw = attw + (((size_t)b * NH_ + h) * S_ + q0) * (size_t)S_;

  // --- Q fragments (contraction d: [0,32) and [32,64)) ---
  v16bf qf[2];
  {
    const bf16_t* qrow = Q + (size_t)(b * S_ + q0 + mlane) * (NH_ * HD_) + h * HD_;
#pragma unroll
    for (int t = 0; t < 2; t++) {
      v8bf a0 = *(const v8bf*)(qrow + t * 32 + hi * 8);
      v8bf a1 = *(const v8bf*)(qrow + t * 32 + 16 + hi * 8);
      qf[t] = SHUF16(a0, a1);
    }
  }

  // --- phase 1: logits ---
  for (int kt = 0; kt < kend; kt += 16) {
    v8f c = {};
    const bf16_t* krow = K + (size_t)(b * S_ + kt + mlane) * (NKV_ * HD_) + kh * HD_;
    __builtin_prefetch(krow + 16 * (NKV_ * HD_), 0, 1);   // next k tile
#pragma unroll
    for (int t = 0; t < 2; t++) {
      v8bf b0 = *(const v8bf*)(krow + t * 32 + hi * 8);
      v8bf b1 = *(const v8bf*)(krow + t * 32 + 16 + hi * 8);
      v16bf bfrag = SHUF16(b0, b1);
      c = __builtin_amdgcn_wmma_f32_16x16x32_bf16(
          false, qf[t], false, bfrag, (short)0, c, false, false);
    }
#pragma unroll
    for (int r = 0; r < 8; r++) {
      int m    = r + hi * 8;
      int kpos = kt + mlane;
      float lg = c[r] * SCALING_;
      if (kpos > q0 + m) lg = NEG_INF_;
      aw[(size_t)m * S_ + kpos] = lg;
    }
  }
  __threadfence_block();   // waits storecnt: logits visible to our own loads

  // --- phase 2: softmax (lane pair (mlane, mlane+16) shares row mlane) ---
  {
    float* row = aw + (size_t)mlane * S_;
    int half = kend >> 1;
    int kbeg = hi * half, klim = kbeg + half;
    float mx = -3.4e38f;
    for (int k = kbeg; k < klim; k++) mx = fmaxf(mx, row[k]);
    mx = fmaxf(mx, __shfl_xor(mx, 16, 32));
    float sum = 0.f;
    for (int k = kbeg; k < klim; k++) sum += __expf(row[k] - mx);
    sum += __shfl_xor(sum, 16, 32);
    float inv = 1.f / sum;
    for (int k = kbeg; k < klim; k++) row[k] = __expf(row[k] - mx) * inv;
    for (int k = kend + hi; k < S_; k += 2) row[k] = 0.f;  // masked tail -> 0
  }
  __threadfence_block();

  // --- phase 3: attn_out = P · V ---
  v8f oacc[4] = {};
  for (int ks = 0; ks < kend; ks += 32) {
    // stage 32 V rows (this kv head's 64 dims) via async DMA; make sure the
    // previous tile's LDS reads retired before the DMA lands on them.
    wait_ds0();
    {
      const bf16_t* vrow = V + (size_t)(b * S_ + ks + lane) * (NKV_ * HD_) + kh * HD_;
      unsigned lv = lds_addr32(&Vt[lane][0]);
#pragma unroll
      for (int c8 = 0; c8 < 8; c8++)
        async_copy_b128(lv + c8 * 16, vrow + c8 * 8);
    }
    wait_async0();   // single wave: DMA complete -> LDS readable

    // A fragment from the probabilities we just wrote (fp32 -> bf16)
    const float* prow = aw + (size_t)mlane * S_ + ks;
    v16bf pf;
#pragma unroll
    for (int i = 0; i < 8; i++) pf[i]     = (bf16_t)prow[hi * 8 + i];
#pragma unroll
    for (int i = 0; i < 8; i++) pf[8 + i] = (bf16_t)prow[16 + hi * 8 + i];

#pragma unroll
    for (int nt = 0; nt < 4; nt++) {
      v16bf vf;
#pragma unroll
      for (int i = 0; i < 16; i++)
        vf[i] = Vt[(i < 8 ? hi * 8 + i : 8 + hi * 8 + i)][nt * 16 + mlane];
      oacc[nt] = __builtin_amdgcn_wmma_f32_16x16x32_bf16(
          false, pf, false, vf, (short)0, oacc[nt], false, false);
    }
  }

#pragma unroll
  for (int nt = 0; nt < 4; nt++)
#pragma unroll
    for (int r = 0; r < 8; r++) {
      int m = r + hi * 8;
      AO[(size_t)(b * S_ + q0 + m) * (NH_ * HD_) + h * HD_ + nt * 16 + mlane] =
          (bf16_t)oacc[nt][r];
    }
}

// ---------------------------------------------------------------------------
// host-side launch
// ---------------------------------------------------------------------------
extern "C" void kernel_launch(void* const* d_in, const int* in_sizes, int n_in,
                              void* d_out, int out_size, void* d_ws, size_t ws_size,
                              hipStream_t stream) {
  (void)in_sizes; (void)n_in; (void)out_size; (void)ws_size;

  const float* Xf   = (const float*)d_in[0];  // hidden_states [B,S,HID]
  const float* cosf = (const float*)d_in[1];  // [B,S,HD]
  const float* sinf = (const float*)d_in[2];  // [B,S,HD]
  // d_in[3] = attention_mask (causal, regenerated on device)
  const float* Wqf  = (const float*)d_in[4];  // [2048,2048]
  const float* Wkf  = (const float*)d_in[5];  // [512,2048]
  const float* Wvf  = (const float*)d_in[6];  // [512,2048]
  const float* Wof  = (const float*)d_in[7];  // [2048,2048]

  float* out  = (float*)d_out;                       // [B,S,HID]
  float* attw = out + (size_t)B_ * S_ * HID_;        // [B,NH,S,S]

  // workspace layout (bf16), total ~76 MB
  constexpr size_t NX  = (size_t)B_ * S_ * HID_;     // 8,388,608
  constexpr size_t NWQ = (size_t)NH_ * HD_ * HID_;   // 4,194,304
  constexpr size_t NWK = (size_t)NKV_ * HD_ * HID_;  // 1,048,576
  constexpr size_t NQ  = (size_t)B_ * S_ * NH_ * HD_;
  constexpr size_t NK  = (size_t)B_ * S_ * NKV_ * HD_;

  bf16_t* Xb  = (bf16_t*)d_ws;
  bf16_t* Wqb = Xb  + NX;
  bf16_t* Wkb = Wqb + NWQ;
  bf16_t* Wvb = Wkb + NWK;
  bf16_t* Wob = Wvb + NWK;
  bf16_t* Qb  = Wob + NWQ;
  bf16_t* Kb  = Qb  + NQ;
  bf16_t* Vb  = Kb  + NK;
  bf16_t* AOb = Vb  + NK;

  const int CT = 256;
  cast_f32_bf16<<<(int)(NX  / CT), CT, 0, stream>>>(Xf,  Xb,  (int)NX);
  cast_f32_bf16<<<(int)(NWQ / CT), CT, 0, stream>>>(Wqf, Wqb, (int)NWQ);
  cast_f32_bf16<<<(int)(NWK / CT), CT, 0, stream>>>(Wkf, Wkb, (int)NWK);
  cast_f32_bf16<<<(int)(NWK / CT), CT, 0, stream>>>(Wvf, Wvb, (int)NWK);
  cast_f32_bf16<<<(int)(NWQ / CT), CT, 0, stream>>>(Wof, Wob, (int)NWQ);

  const int M = B_ * S_;                             // 4096 tokens
  // QKV projections
  gemm_nt_bf16<true><<<dim3(M / 128, (NH_ * HD_) / 128), 256, 0, stream>>>(
      Xb, Wqb, Qb, M, NH_ * HD_, HID_);
  gemm_nt_bf16<true><<<dim3(M / 128, (NKV_ * HD_) / 128), 256, 0, stream>>>(
      Xb, Wkb, Kb, M, NKV_ * HD_, HID_);
  gemm_nt_bf16<true><<<dim3(M / 128, (NKV_ * HD_) / 128), 256, 0, stream>>>(
      Xb, Wvb, Vb, M, NKV_ * HD_, HID_);

  // RoPE
  {
    int tq = B_ * S_ * NH_ * 32;
    rope_kernel<<<tq / CT, CT, 0, stream>>>(Qb, cosf, sinf, NH_);
    int tk = B_ * S_ * NKV_ * 32;
    rope_kernel<<<tk / CT, CT, 0, stream>>>(Kb, cosf, sinf, NKV_);
  }

  // attention: softmax weights -> d_out (fp32), attn_out -> AOb (bf16)
  attn_kernel<<<dim3(S_ / 16, NH_, B_), 32, 0, stream>>>(Qb, Kb, Vb, attw, AOb);

  // output projection -> d_out (fp32)
  gemm_nt_bf16<false><<<dim3(M / 128, HID_ / 128), 256, 0, stream>>>(
      AOb, Wob, out, M, HID_, HID_);
}
